// modelf_90177133347146
// MI455X (gfx1250) — matrix operand order repaired and checked by reading, not compiled, not run on hardware
//
#include <hip/hip_runtime.h>

// ---------------------------------------------------------------------------
// CDNA5 (gfx1250) seq2seq attention model.
// Precision strategy: f16 WMMA (v_wmma_f32_16x16x32_f16) with f32 accumulate.
// Weights repacked once into fragment-major f16 so B operands are one 32B
// load per lane; activations staged in LDS; encoder recurrence is per-batch-
// tile private (no grid sync).
// ---------------------------------------------------------------------------

typedef _Float16 half_t;
typedef __attribute__((ext_vector_type(16))) _Float16 v16h;
typedef __attribute__((ext_vector_type(8)))  float    v8f;
typedef __attribute__((ext_vector_type(4)))  int      i4x;   // 16 bytes
typedef __attribute__((ext_vector_type(8)))  int      i8x;   // 32 bytes

union FragU {
  v16h h;
  i4x  q[2];
  i8x  o;
};

// A fragment (16x32 f16): lane L<16 -> row L, K = {0..7, 16..23};
// lane L>=16 -> row L-16, K = {8..15, 24..31}. p points at row start + k0.
__device__ inline v16h load_a(const half_t* p, int lane) {
  const int kb = (lane & 16) ? 8 : 0;
  FragU f;
  f.q[0] = *(const i4x*)(p + kb);
  f.q[1] = *(const i4x*)(p + kb + 16);
  return f.h;
}

// B fragment (32x16 f16) from pre-swizzled storage: 512 f16 per fragment,
// lane's 16 halves contiguous at frag + lane*16.
__device__ inline v16h load_b(const half_t* frag, int lane) {
  FragU f;
  f.o = *(const i8x*)(frag + lane * 16);
  return f.h;
}

__device__ inline v8f wmma_f16(v16h a, v16h b, v8f c) {
  return __builtin_amdgcn_wmma_f32_16x16x32_f16(false, a, false, b,
                                                (short)0, c, false, false);
}

__device__ inline float sigm(float x) { return 1.0f / (1.0f + __expf(-x)); }

// ---------------------------------------------------------------------------
// Repack W (OUT x K, row-major f32) into fragment-major f16 for B operands.
// Fragment (kt, nt) holds B[k, n] = W[nt*16+n][kt*32+k].
// B VGPR layout: lanes 0-15 hold K = 0..7,16..23 (2 per VGPR), lanes 16-31 K=8..15,24..31.
// One block (256 threads) packs one fragment (32 lanes x 8 pairs).
// ---------------------------------------------------------------------------
__global__ __launch_bounds__(256) void pack_kernel(
    const float* __restrict__ W, half_t* __restrict__ dst,
    int OUT, int K, int NT) {
  const int frag = blockIdx.x;
  const int kt = frag / NT;
  const int nt = frag % NT;
  const int tid = threadIdx.x;
  const int lane = tid >> 3;          // 0..31
  const int v = tid & 7;              // 0..7
  const int n = lane & 15;
  const int kh = (lane & 16) ? 8 : 0;
  const int k0 = kt * 32 + kh + v * 2 + ((v & 4) ? 8 : 0);
  const int row = nt * 16 + n;
  half_t a0 = (half_t)0.f, a1 = (half_t)0.f;
  if (row < OUT) {
    a0 = (half_t)W[(size_t)row * K + k0];
    a1 = (half_t)W[(size_t)row * K + k0 + 1];
  }
  half_t* d = dst + (size_t)frag * 512 + lane * 16 + v * 2;
  d[0] = a0;
  d[1] = a1;
}

// ---------------------------------------------------------------------------
// Encoder: grid (16 batch tiles, 2 directions), 256 threads (8 waves).
// Each block runs the full 256-step LSTM recurrence for its 16 batch rows.
// Per step: g(16x1024) = x_t @ Wih^T + h @ Whh^T  via WMMA, then gate math.
// Dynamic LDS: hBuf 8K, xBuf 4K, gBuf 64K, cBuf 16K  (= 94208 B).
// ---------------------------------------------------------------------------
__global__ __launch_bounds__(256) void enc_kernel(
    const float* __restrict__ X,       // (256, 256, 128)
    const half_t* __restrict__ pWihF, const half_t* __restrict__ pWhhF,
    const float* __restrict__ bF,
    const half_t* __restrict__ pWihB, const half_t* __restrict__ pWhhB,
    const float* __restrict__ bB,
    half_t* __restrict__ aOut) {       // (256, 256, 512) f16
  extern __shared__ __align__(16) char smem[];
  half_t* hBuf = (half_t*)smem;                  //  16*256 f16
  half_t* xBuf = (half_t*)(smem + 8192);         //  16*128 f16
  float*  gBuf = (float*)(smem + 12288);         //  16*1024 f32
  float*  cBuf = (float*)(smem + 77824);         //  16*256 f32

  const int bt  = blockIdx.x;                    // batch tile 0..15
  const int dir = blockIdx.y;                    // 0=fwd, 1=bwd
  const half_t* pWih = dir ? pWihB : pWihF;
  const half_t* pWhh = dir ? pWhhB : pWhhF;
  const float*  bias = dir ? bB : bF;

  const int tid = threadIdx.x;
  const int wid = tid >> 5;
  const int lane = tid & 31;

  for (int i = tid; i < 16 * 256; i += 256) { hBuf[i] = (half_t)0.f; cBuf[i] = 0.f; }
  __syncthreads();

  for (int t = 0; t < 256; ++t) {
    const int time = dir ? (255 - t) : t;
    // stage x_t tile (16 x 128) as f16
    for (int i = tid; i < 16 * 128; i += 256) {
      const int row = i >> 7, col = i & 127;
      xBuf[i] = (half_t)X[(((size_t)(bt * 16 + row)) * 256 + time) * 128 + col];
    }
    __syncthreads();

    // GEMM: each wave handles 8 of 64 N-tiles; K = 128 (x) + 256 (h)
    const half_t* aRowX = xBuf + (lane & 15) * 128;
    const half_t* aRowH = hBuf + (lane & 15) * 256;
    for (int q = 0; q < 8; ++q) {
      const int nt = wid * 8 + q;
      v8f acc = {};
      for (int kt = 0; kt < 4; ++kt)
        acc = wmma_f16(load_a(aRowX + kt * 32, lane),
                       load_b(pWih + ((size_t)(kt * 64 + nt)) * 512, lane), acc);
      for (int kt = 0; kt < 8; ++kt)
        acc = wmma_f16(load_a(aRowH + kt * 32, lane),
                       load_b(pWhh + ((size_t)(kt * 64 + nt)) * 512, lane), acc);
      const int n = lane & 15;
      const int rbase = (lane >> 4) << 3;
      for (int v = 0; v < 8; ++v)
        gBuf[(rbase + v) * 1024 + nt * 16 + n] = acc[v];
    }
    __syncthreads();

    // gate math: i,f,g,o slices of 256; update c, h; write a
    for (int idx = tid; idx < 16 * 256; idx += 256) {
      const int row = idx >> 8, j = idx & 255;
      const int gb = row * 1024;
      const float ig = gBuf[gb + j]       + bias[j];
      const float fg = gBuf[gb + 256 + j] + bias[256 + j];
      const float gg = gBuf[gb + 512 + j] + bias[512 + j];
      const float og = gBuf[gb + 768 + j] + bias[768 + j];
      float c = sigm(fg) * cBuf[idx] + sigm(ig) * tanhf(gg);
      float h = sigm(og) * tanhf(c);
      cBuf[idx] = c;
      hBuf[idx] = (half_t)h;
      aOut[(((size_t)(bt * 16 + row)) * 256 + time) * 512 + dir * 256 + j] = (half_t)h;
    }
    __syncthreads();
  }
}

// ---------------------------------------------------------------------------
// Attention scores: e = tanh([a, s] @ W1^T + b1); score = relu(e @ W2^T + b2).
// Grid 512 (= 256 batch x 2 halves of Tx); 8 waves, each owns 16 tx rows.
// One 16x16 WMMA tile over K=1024 (first 512 from a, last 512 from s).
// ---------------------------------------------------------------------------
__global__ __launch_bounds__(256) void attn_kernel(
    const half_t* __restrict__ aF16,   // (256, 256, 512)
    const half_t* __restrict__ sF16,   // (256, 512)
    const half_t* __restrict__ pW1,    // 32 fragments (K=1024, NT=1 padded)
    const float* __restrict__ b1, const float* __restrict__ W2,
    const float* __restrict__ b2,
    float* __restrict__ score) {       // (256, 256)
  __shared__ __align__(16) half_t sRow[512];
  const int b = blockIdx.x >> 1;
  const int txBase = (blockIdx.x & 1) * 128;
  const int tid = threadIdx.x;
  const int wid = tid >> 5, lane = tid & 31;

  for (int i = tid; i < 512; i += 256) sRow[i] = sF16[b * 512 + i];
  __syncthreads();

  const int tx = txBase + wid * 16 + (lane & 15);
  const half_t* aRow = aF16 + (((size_t)b * 256) + tx) * 512;

  v8f acc = {};
  for (int kt = 0; kt < 32; ++kt) {
    const int k0 = kt * 32;
    v16h A;
    if (k0 < 512) A = load_a(aRow + k0, lane);
    else          A = load_a(sRow + (k0 - 512), lane);
    acc = wmma_f16(A, load_b(pW1 + (size_t)kt * 512, lane), acc);
  }

  const int n = lane & 15;
  const float w2n = (n < 10) ? W2[n] : 0.f;
  const float b1n = (n < 10) ? b1[n] : 0.f;
  const float b2v = b2[0];
  const int rbase = (lane >> 4) << 3;
  for (int v = 0; v < 8; ++v) {
    float e = tanhf(acc[v] + b1n) * w2n;
    for (int m = 1; m < 16; m <<= 1) e += __shfl_xor(e, m, 32);
    if (n == 0) {
      const int row = txBase + wid * 16 + rbase + v;
      score[b * 256 + row] = e + b2v > 0.f ? e + b2v : 0.f;
    }
  }
}

// ---------------------------------------------------------------------------
// Softmax over Tx + context = sum(alpha * a). One block per batch row.
// ---------------------------------------------------------------------------
__global__ __launch_bounds__(256) void smax_ctx_kernel(
    const float* __restrict__ score, const half_t* __restrict__ aF16,
    half_t* __restrict__ ctxF16) {     // (256, 512)
  __shared__ float sc[256];
  __shared__ float red[256];
  const int b = blockIdx.x, tid = threadIdx.x;
  const float v = score[b * 256 + tid];
  sc[tid] = v; red[tid] = v;
  __syncthreads();
  for (int s = 128; s > 0; s >>= 1) {
    if (tid < s) red[tid] = fmaxf(red[tid], red[tid + s]);
    __syncthreads();
  }
  const float mx = red[0];
  __syncthreads();
  const float e = __expf(v - mx);
  sc[tid] = e; red[tid] = e;
  __syncthreads();
  for (int s = 128; s > 0; s >>= 1) {
    if (tid < s) red[tid] += red[tid + s];
    __syncthreads();
  }
  const float inv = 1.f / red[0];
  __syncthreads();

  const half_t* aRow = aF16 + (size_t)b * 256 * 512;
  const int d0 = tid * 2;
  float a0 = 0.f, a1 = 0.f;
  for (int tx = 0; tx < 256; ++tx) {
    const float al = sc[tx];
    a0 += al * (float)aRow[tx * 512 + d0];
    a1 += al * (float)aRow[tx * 512 + d0 + 1];
  }
  ctxF16[b * 512 + d0]     = (half_t)(a0 * inv);
  ctxF16[b * 512 + d0 + 1] = (half_t)(a1 * inv);
}

// ---------------------------------------------------------------------------
// Decoder LSTMCell: g = ctx @ Wc_ih^T + s @ Wc_hh^T + bc; update c, s.
// Grid 16 (batch tiles); 8 waves x 16 N-tiles each (out = 2048).
// Dynamic LDS: ctxL 16K + sL 16K + gBuf 128K = 163840 B.
// ---------------------------------------------------------------------------
__global__ __launch_bounds__(256) void dec_cell_kernel(
    const half_t* __restrict__ ctxF16,
    half_t* __restrict__ sF16,          // read old, write new (per-row private)
    const half_t* __restrict__ pWcih, const half_t* __restrict__ pWchh,
    const float* __restrict__ bc,
    float* __restrict__ cState) {       // (256, 512)
  extern __shared__ __align__(16) char smem[];
  half_t* ctxL = (half_t*)smem;                   // 16*512 f16
  half_t* sL   = (half_t*)(smem + 16384);         // 16*512 f16
  float*  gBuf = (float*)(smem + 32768);          // 16*2048 f32

  const int bt = blockIdx.x;
  const int tid = threadIdx.x;
  const int wid = tid >> 5, lane = tid & 31;

  for (int i = tid; i < 16 * 512; i += 256) {
    const int row = i >> 9, j = i & 511;
    ctxL[i] = ctxF16[(size_t)(bt * 16 + row) * 512 + j];
    sL[i]   = sF16[(size_t)(bt * 16 + row) * 512 + j];
  }
  __syncthreads();

  const half_t* aRowC = ctxL + (lane & 15) * 512;
  const half_t* aRowS = sL + (lane & 15) * 512;
  for (int q = 0; q < 16; ++q) {
    const int nt = wid * 16 + q;
    v8f acc = {};
    for (int kt = 0; kt < 16; ++kt)
      acc = wmma_f16(load_a(aRowC + kt * 32, lane),
                     load_b(pWcih + ((size_t)(kt * 128 + nt)) * 512, lane), acc);
    for (int kt = 0; kt < 16; ++kt)
      acc = wmma_f16(load_a(aRowS + kt * 32, lane),
                     load_b(pWchh + ((size_t)(kt * 128 + nt)) * 512, lane), acc);
    const int n = lane & 15;
    const int rbase = (lane >> 4) << 3;
    for (int v = 0; v < 8; ++v)
      gBuf[(rbase + v) * 2048 + nt * 16 + n] = acc[v];
  }
  __syncthreads();

  for (int idx = tid; idx < 16 * 512; idx += 256) {
    const int row = idx >> 9, j = idx & 511;
    const int gb = row * 2048;
    const float ig = gBuf[gb + j]        + bc[j];
    const float fg = gBuf[gb + 512 + j]  + bc[512 + j];
    const float gg = gBuf[gb + 1024 + j] + bc[1024 + j];
    const float og = gBuf[gb + 1536 + j] + bc[1536 + j];
    const size_t gi = (size_t)(bt * 16 + row) * 512 + j;
    float c = sigm(fg) * cState[gi] + sigm(ig) * tanhf(gg);
    float s = sigm(og) * tanhf(c);
    cState[gi] = c;
    sF16[gi] = (half_t)s;
  }
}

// ---------------------------------------------------------------------------
// Output projection: out[:, t, :] = s @ Wfc^T + bfc. Grid 64 = 16 tiles x 4.
// ---------------------------------------------------------------------------
__global__ __launch_bounds__(256) void out_kernel(
    const half_t* __restrict__ sF16, const half_t* __restrict__ pWfc,
    const float* __restrict__ bfc, float* __restrict__ out, int t) {
  __shared__ __align__(16) half_t sL[16 * 512];
  const int bt = blockIdx.x >> 2;
  const int ng = blockIdx.x & 3;
  const int tid = threadIdx.x;
  const int wid = tid >> 5, lane = tid & 31;

  for (int i = tid; i < 16 * 512; i += 256) {
    const int row = i >> 9, j = i & 511;
    sL[i] = sF16[(size_t)(bt * 16 + row) * 512 + j];
  }
  __syncthreads();

  const half_t* aRow = sL + (lane & 15) * 512;
  for (int q = 0; q < 2; ++q) {
    const int nt = ng * 16 + wid * 2 + q;
    v8f acc = {};
    for (int kt = 0; kt < 16; ++kt)
      acc = wmma_f16(load_a(aRow + kt * 32, lane),
                     load_b(pWfc + ((size_t)(kt * 64 + nt)) * 512, lane), acc);
    const int n = lane & 15;
    const int col = nt * 16 + n;
    const float bv = bfc[col];
    const int rbase = (lane >> 4) << 3;
    for (int v = 0; v < 8; ++v) {
      const int gRow = bt * 16 + rbase + v;
      out[((size_t)gRow * 10 + t) * 1024 + col] = acc[v] + bv;
    }
  }
}

// ---------------------------------------------------------------------------
extern "C" void kernel_launch(void* const* d_in, const int* in_sizes, int n_in,
                              void* d_out, int out_size, void* d_ws, size_t ws_size,
                              hipStream_t stream) {
  (void)in_sizes; (void)n_in; (void)out_size; (void)ws_size;
  const float* X     = (const float*)d_in[0];
  const float* Wih_f = (const float*)d_in[1];
  const float* Whh_f = (const float*)d_in[2];
  const float* b_f   = (const float*)d_in[3];
  const float* Wih_b = (const float*)d_in[4];
  const float* Whh_b = (const float*)d_in[5];
  const float* b_b   = (const float*)d_in[6];
  const float* W1    = (const float*)d_in[7];
  const float* b1    = (const float*)d_in[8];
  const float* W2    = (const float*)d_in[9];
  const float* b2    = (const float*)d_in[10];
  const float* Wc_ih = (const float*)d_in[11];
  const float* Wc_hh = (const float*)d_in[12];
  const float* bc    = (const float*)d_in[13];
  const float* Wfc   = (const float*)d_in[14];
  const float* bfc   = (const float*)d_in[15];
  float* out = (float*)d_out;

  char* ws = (char*)d_ws;
  size_t off = 0;
  auto alloc = [&](size_t bytes) -> char* {
    char* p = ws + off;
    off = (off + bytes + 255) & ~(size_t)255;
    return p;
  };
  half_t* aF16   = (half_t*)alloc((size_t)256 * 256 * 512 * 2);  // 64 MB
  half_t* pWihF  = (half_t*)alloc((size_t)4  * 64  * 1024);      // 256 KB
  half_t* pWhhF  = (half_t*)alloc((size_t)8  * 64  * 1024);      // 512 KB
  half_t* pWihB  = (half_t*)alloc((size_t)4  * 64  * 1024);
  half_t* pWhhB  = (half_t*)alloc((size_t)8  * 64  * 1024);
  half_t* pW1    = (half_t*)alloc((size_t)32 * 1   * 1024);      // 32 KB
  half_t* pWcih  = (half_t*)alloc((size_t)16 * 128 * 1024);      // 2 MB
  half_t* pWchh  = (half_t*)alloc((size_t)16 * 128 * 1024);      // 2 MB
  half_t* pWfc   = (half_t*)alloc((size_t)16 * 64  * 1024);      // 1 MB
  half_t* sF16   = (half_t*)alloc((size_t)256 * 512 * 2);        // 256 KB
  float*  cF32   = (float*) alloc((size_t)256 * 512 * 4);        // 512 KB
  half_t* ctxF16 = (half_t*)alloc((size_t)256 * 512 * 2);        // 256 KB
  float*  score  = (float*) alloc((size_t)256 * 256 * 4);        // 256 KB

  // Repack weights into fragment-major f16 (L2-resident thereafter).
  pack_kernel<<<4 * 64,   256, 0, stream>>>(Wih_f, pWihF, 1024, 128, 64);
  pack_kernel<<<8 * 64,   256, 0, stream>>>(Whh_f, pWhhF, 1024, 256, 64);
  pack_kernel<<<4 * 64,   256, 0, stream>>>(Wih_b, pWihB, 1024, 128, 64);
  pack_kernel<<<8 * 64,   256, 0, stream>>>(Whh_b, pWhhB, 1024, 256, 64);
  pack_kernel<<<32 * 1,   256, 0, stream>>>(W1,    pW1,   10,  1024, 1);
  pack_kernel<<<16 * 128, 256, 0, stream>>>(Wc_ih, pWcih, 2048, 512, 128);
  pack_kernel<<<16 * 128, 256, 0, stream>>>(Wc_hh, pWchh, 2048, 512, 128);
  pack_kernel<<<16 * 64,  256, 0, stream>>>(Wfc,   pWfc,  1024, 512, 64);

  // Zero decoder state (graph-capturable async memsets).
  hipMemsetAsync(sF16, 0, (size_t)256 * 512 * 2, stream);
  hipMemsetAsync(cF32, 0, (size_t)256 * 512 * 4, stream);

  // Bidirectional encoder: 16 batch tiles x 2 directions, private recurrence.
  enc_kernel<<<dim3(16, 2, 1), 256, 94208, stream>>>(
      X, pWihF, pWhhF, b_f, pWihB, pWhhB, b_b, aF16);

  // Decoder: 10 steps of attention -> softmax/context -> LSTMCell -> vocab.
  for (int t = 0; t < 10; ++t) {
    attn_kernel<<<512, 256, 0, stream>>>(aF16, sF16, pW1, b1, W2, b2, score);
    smax_ctx_kernel<<<256, 256, 0, stream>>>(score, aF16, ctxF16);
    dec_cell_kernel<<<16, 256, 163840, stream>>>(ctxF16, sF16, pWcih, pWchh, bc, cF32);
    out_kernel<<<64, 256, 0, stream>>>(sF16, pWfc, bfc, out, t);
  }
}
